// MultiHeadAttn_26070451486894
// MI455X (gfx1250) — matrix operand order repaired and checked by reading, not compiled, not run on hardware
//
#include <hip/hip_runtime.h>

#define BB 4
#define NN 512
#define MM 2048
#define HID 512
#define HH 16
#define KD 64
#define HK 1024

typedef __attribute__((ext_vector_type(16))) _Float16 v16h;
typedef __attribute__((ext_vector_type(8)))  _Float16 v8h;
typedef __attribute__((ext_vector_type(8)))  float    v8f;

// A-fragment from f16 memory: two contiguous 8-half runs -> halves [0..7],[8..15].
__device__ __forceinline__ v16h join8(const _Float16* __restrict__ p0,
                                      const _Float16* __restrict__ p1) {
  v8h a = *(const v8h*)p0;
  v8h b = *(const v8h*)p1;
  v16h r;
#pragma unroll
  for (int i = 0; i < 8; i++) { r[i] = a[i]; r[8 + i] = b[i]; }
  return r;
}

#define WMMA16(A, Bf, C) \
  __builtin_amdgcn_wmma_f32_16x16x32_f16(false, (A), false, (Bf), (short)0, (C), false, false)

// ---------------------------------------------------------------------------
// One-shot f32 -> f16 conversion into workspace (8 elements / thread).
// ---------------------------------------------------------------------------
__global__ __launch_bounds__(256) void cvt_kernel(const float* __restrict__ src,
                                                  _Float16* __restrict__ dst,
                                                  int n) {
  int i = (blockIdx.x * 256 + threadIdx.x) * 8;
  if (i >= n) return;
  float4 a = *(const float4*)(src + i);
  float4 b = *(const float4*)(src + i + 4);
  v8h o;
  o[0] = (_Float16)a.x; o[1] = (_Float16)a.y; o[2] = (_Float16)a.z; o[3] = (_Float16)a.w;
  o[4] = (_Float16)b.x; o[5] = (_Float16)b.y; o[6] = (_Float16)b.z; o[7] = (_Float16)b.w;
  *(v8h*)(dst + i) = o;
}

// ---------------------------------------------------------------------------
// Phase A: C[row,col] = X16[row,:] @ W16[col,:] + bias[col]; optional RoPE.
// Waves within a block share one head (-> shared weight slice in WGP$/L2) and
// take consecutive 16-row tiles.
// mode 0: RoPE + row-major f16 out[(b*H+h)*rowsPerB + pos][64]
// mode 1: no RoPE, transposed f16 out[((b*H+h)*64 + kv)*rowsPerB + pos]
// ---------------------------------------------------------------------------
__global__ __launch_bounds__(256) void proj_kernel(
    const _Float16* __restrict__ X16, const _Float16* __restrict__ W16,
    const float* __restrict__ bias, const float* __restrict__ rope,
    _Float16* __restrict__ out, int rowsPerB, int nRT, int mode) {
  __shared__ float lds[8 * 16 * 64];
  const int tid = threadIdx.x, lane = tid & 31, wv = tid >> 5;
  const int ln = lane & 15, g2 = lane >> 4;
  const int waveId = blockIdx.x * 8 + wv;
  const int h = waveId / nRT;          // waves in a block -> same head
  const int rowtile = waveId - h * nRT;
  const int row0 = rowtile * 16;
  const int b = row0 / rowsPerB;
  const int p0 = row0 % rowsPerB;
  const int colbase = h * 64;
  const _Float16* xrow = X16 + (size_t)(row0 + ln) * HID;

  v8f acc[4] = {{}, {}, {}, {}};
#pragma unroll 1
  for (int k = 0; k < HID; k += 32) {
    v16h a = join8(xrow + k + 16 * g2, xrow + k + 16 * g2 + 8);
#pragma unroll
    for (int t = 0; t < 4; t++) {
      const _Float16* wp = W16 + (size_t)(colbase + t * 16 + ln) * HID + k + 16 * g2;
      v16h bf = *(const v16h*)wp;
      acc[t] = WMMA16(a, bf, acc[t]);
    }
  }
#pragma unroll
  for (int t = 0; t < 4; t++) {
    float bc = bias[colbase + t * 16 + ln];
#pragma unroll
    for (int r = 0; r < 8; r++) acc[t][r] += bc;
  }

  if (mode == 1) {  // V: store transposed, rows are consecutive per lane
#pragma unroll
    for (int t = 0; t < 4; t++) {
      v8h vv;
#pragma unroll
      for (int r = 0; r < 8; r++) vv[r] = (_Float16)acc[t][r];
      _Float16* vp = out + ((size_t)(b * HH + h) * 64 + t * 16 + ln) * rowsPerB
                     + p0 + 8 * g2;
      *(v8h*)vp = vv;
    }
  } else {  // Q/K: stage tile in LDS, apply RoPE, store row-major f16
    float* lb = lds + wv * 1024;
#pragma unroll
    for (int t = 0; t < 4; t++)
#pragma unroll
      for (int r = 0; r < 8; r++)
        lb[(r + 8 * g2) * 64 + t * 16 + ln] = acc[t][r];
    __syncthreads();
    const int i0 = g2 * 16;  // each half-wave handles 16 of the 32 rope pairs
    const float* xr = lb + ln * 64;
    const float* dis = rope + ((size_t)(row0 + ln) * HH + h) * 64;
    v8h o0a, o0b, o1a, o1b;
#pragma unroll
    for (int u = 0; u < 16; u++) {
      int i = i0 + u;
      float2 xp = *(const float2*)(xr + 2 * i);
      float sn = dis[i], cs = dis[32 + i];
      float e0 = xp.x * cs - xp.y * sn;
      float e1 = xp.y * cs + xp.x * sn;
      if (u < 8) { o0a[u] = (_Float16)e0; o1a[u] = (_Float16)e1; }
      else       { o0b[u - 8] = (_Float16)e0; o1b[u - 8] = (_Float16)e1; }
    }
    _Float16* orow = out + ((size_t)(b * HH + h) * rowsPerB + p0 + ln) * 64;
    *(v8h*)(orow + i0) = o0a;
    *(v8h*)(orow + i0 + 8) = o0b;
    *(v8h*)(orow + 32 + i0) = o1a;
    *(v8h*)(orow + 32 + i0 + 8) = o1b;
  }
}

// ---------------------------------------------------------------------------
// Phase B: flash attention per (b,h, 16-row n-tile). Computes S^T = K·Q^T so
// exp'd scores land directly in the A-fragment layout for P@V. The 4 waves of
// a block share one (b,h) K/V stream (consecutive n-tiles) for cache reuse.
// ---------------------------------------------------------------------------
__global__ __launch_bounds__(128) void attn_kernel(
    const _Float16* __restrict__ qh, const _Float16* __restrict__ kh,
    const _Float16* __restrict__ vt, const unsigned char* __restrict__ mask,
    _Float16* __restrict__ px) {
  const int tid = threadIdx.x, lane = tid & 31, wv = tid >> 5;
  const int ln = lane & 15, g2 = lane >> 4;
  const int waveId = blockIdx.x * 4 + wv;
  const int ntile = waveId & 31;      // N/16
  const int bh = waveId >> 5;         // b*16 + h
  const int h = bh & 15, b = bh >> 4;
  const int n0 = ntile * 16;

  const _Float16* qrow = qh + ((size_t)bh * NN + n0 + ln) * 64;
  const v16h qb0 = join8(qrow + 8 * g2, qrow + 16 + 8 * g2);        // kk 0..31
  const v16h qb1 = join8(qrow + 32 + 8 * g2, qrow + 48 + 8 * g2);   // kk 32..63
  const _Float16* kb = kh + (size_t)bh * MM * 64;
  const _Float16* vbas = vt + (size_t)bh * 64 * MM;
  const unsigned char* mrow = mask + ((size_t)bh * NN + n0 + ln) * MM;

  float m_run = -3.0e38f, s_run = 0.0f;
  v8f o0 = {}, o1 = {}, o2 = {}, o3 = {};

#pragma unroll 1
  for (int mc = 0; mc < MM; mc += 32) {
    const _Float16* kr0 = kb + (size_t)(mc + ln) * 64;
    const _Float16* kr1 = kb + (size_t)(mc + 16 + ln) * 64;
    if (mc + 32 < MM) {
      __builtin_prefetch(kr0 + 32 * 64, 0, 1);
      __builtin_prefetch(mrow + mc + 32, 0, 1);
    }
    v8f c0 = {}, c1 = {};
    {
      v16h a = join8(kr0 + 8 * g2, kr0 + 16 + 8 * g2);
      c0 = WMMA16(a, qb0, c0);
      a = join8(kr0 + 32 + 8 * g2, kr0 + 48 + 8 * g2);
      c0 = WMMA16(a, qb1, c0);
      a = join8(kr1 + 8 * g2, kr1 + 16 + 8 * g2);
      c1 = WMMA16(a, qb0, c1);
      a = join8(kr1 + 32 + 8 * g2, kr1 + 48 + 8 * g2);
      c1 = WMMA16(a, qb1, c1);
    }
    unsigned long long mm0 = *(const unsigned long long*)(mrow + mc + 8 * g2);
    unsigned long long mm1 = *(const unsigned long long*)(mrow + mc + 16 + 8 * g2);
#pragma unroll
    for (int r = 0; r < 8; r++) {
      c0[r] = ((mm0 >> (8 * r)) & 0xffull) ? c0[r] * 0.125f : -1.0e30f;
      c1[r] = ((mm1 >> (8 * r)) & 0xffull) ? c1[r] * 0.125f : -1.0e30f;
    }
    float cm = c0[0];
#pragma unroll
    for (int r = 0; r < 8; r++) { cm = fmaxf(cm, c0[r]); cm = fmaxf(cm, c1[r]); }
    cm = fmaxf(cm, __shfl_xor(cm, 16));
    float m_new = fmaxf(m_run, cm);
    float rs = __expf(m_run - m_new);
    float psum = 0.0f;
    v16h pa;
#pragma unroll
    for (int r = 0; r < 8; r++) {
      float p = __expf(c0[r] - m_new); psum += p; pa[r] = (_Float16)p;
    }
#pragma unroll
    for (int r = 0; r < 8; r++) {
      float p = __expf(c1[r] - m_new); psum += p; pa[8 + r] = (_Float16)p;
    }
    psum += __shfl_xor(psum, 16);
    s_run = s_run * rs + psum;
    m_run = m_new;
    // per-row rescale of O accumulators (row n lives at lane n in score layout)
    float sc[8];
#pragma unroll
    for (int r = 0; r < 8; r++) sc[r] = __shfl(rs, r + 8 * g2);
#pragma unroll
    for (int r = 0; r < 8; r++) {
      o0[r] *= sc[r]; o1[r] *= sc[r]; o2[r] *= sc[r]; o3[r] *= sc[r];
    }
    const _Float16* vp = vbas + (size_t)ln * MM + mc;
    v16h vB = join8(vp + 8 * g2, vp + 16 + 8 * g2); o0 = WMMA16(pa, vB, o0);
    vB = join8(vp + (size_t)16 * MM + 8 * g2, vp + (size_t)16 * MM + 16 + 8 * g2); o1 = WMMA16(pa, vB, o1);
    vB = join8(vp + (size_t)32 * MM + 8 * g2, vp + (size_t)32 * MM + 16 + 8 * g2); o2 = WMMA16(pa, vB, o2);
    vB = join8(vp + (size_t)48 * MM + 8 * g2, vp + (size_t)48 * MM + 16 + 8 * g2); o3 = WMMA16(pa, vB, o3);
  }
  float inv = 1.0f / s_run;
  float dv[8];
#pragma unroll
  for (int r = 0; r < 8; r++) dv[r] = __shfl(inv, r + 8 * g2);
#pragma unroll
  for (int r = 0; r < 8; r++) {
    size_t off = (size_t)(b * NN + n0 + r + 8 * g2) * HK + h * 64 + ln;
    px[off + 0]  = (_Float16)(o0[r] * dv[r]);
    px[off + 16] = (_Float16)(o1[r] * dv[r]);
    px[off + 32] = (_Float16)(o2[r] * dv[r]);
    px[off + 48] = (_Float16)(o3[r] * dv[r]);
  }
}

// ---------------------------------------------------------------------------
// Phase C: out = LN(2*tgt + postx @ Wo^T + bo). One block per 16-row tile,
// 8 waves each owning 64 output columns; LDS tile for residual + LayerNorm.
// ---------------------------------------------------------------------------
__global__ __launch_bounds__(256) void out_ln_kernel(
    const _Float16* __restrict__ px, const _Float16* __restrict__ Wo16,
    const float* __restrict__ bo, const float* __restrict__ tgt,
    const float* __restrict__ gamma, const float* __restrict__ beta,
    float* __restrict__ out) {
  __shared__ float xs[16 * 512];
  __shared__ float ps[256], ps2[256];
  __shared__ float mu_s[16], rstd_s[16];
  const int tid = threadIdx.x, lane = tid & 31, wv = tid >> 5;
  const int ln = lane & 15, g2 = lane >> 4;
  const int row0 = blockIdx.x * 16;
  const int colbase = wv * 64;
  const _Float16* arow = px + (size_t)(row0 + ln) * HK;

  v8f acc[4] = {{}, {}, {}, {}};
#pragma unroll 1
  for (int k = 0; k < HK; k += 32) {
    v16h a = join8(arow + k + 16 * g2, arow + k + 16 * g2 + 8);
#pragma unroll
    for (int t = 0; t < 4; t++) {
      const _Float16* wp = Wo16 + (size_t)(colbase + t * 16 + ln) * HK + k + 16 * g2;
      v16h bf = *(const v16h*)wp;
      acc[t] = WMMA16(a, bf, acc[t]);
    }
  }
#pragma unroll
  for (int t = 0; t < 4; t++) {
    float bc = bo[colbase + t * 16 + ln];
#pragma unroll
    for (int r = 0; r < 8; r++)
      xs[(r + 8 * g2) * 512 + colbase + t * 16 + ln] = acc[t][r] + bc;
  }
  __syncthreads();

  const int r = tid >> 4, j = tid & 15;
  const float* trow = tgt + (size_t)(row0 + r) * HID;
  float sum = 0.0f, sq = 0.0f;
#pragma unroll 1
  for (int c = j; c < HID; c += 16) {
    float x = xs[r * 512 + c] + 2.0f * trow[c];  // ALPHA = 2.0
    xs[r * 512 + c] = x;
    sum += x; sq += x * x;
  }
  ps[tid] = sum; ps2[tid] = sq;
  __syncthreads();
  if (tid < 16) {
    float s = 0.0f, q = 0.0f;
#pragma unroll
    for (int i = 0; i < 16; i++) { s += ps[tid * 16 + i]; q += ps2[tid * 16 + i]; }
    float mu = s * (1.0f / HID);
    float var = q * (1.0f / HID) - mu * mu;
    mu_s[tid] = mu;
    rstd_s[tid] = rsqrtf(var + 1e-5f);
  }
  __syncthreads();
  float mu = mu_s[r], rstd = rstd_s[r];
  float* orow = out + (size_t)(row0 + r) * HID;
#pragma unroll 1
  for (int c = j; c < HID; c += 16)
    orow[c] = (xs[r * 512 + c] - mu) * rstd * gamma[c] + beta[c];
}

// ---------------------------------------------------------------------------
extern "C" void kernel_launch(void* const* d_in, const int* in_sizes, int n_in,
                              void* d_out, int out_size, void* d_ws, size_t ws_size,
                              hipStream_t stream) {
  (void)in_sizes; (void)n_in; (void)ws_size;
  const float* tgt = (const float*)d_in[0];
  const float* mem = (const float*)d_in[1];
  const float* gly = (const float*)d_in[2];
  const float* nme = (const float*)d_in[3];
  const unsigned char* mask = (const unsigned char*)d_in[5];
  const void* A = d_in[7];
  const float* Wq = (const float*)d_in[8];
  const float* bq = (const float*)d_in[9];
  const float* Wk = (const float*)d_in[10];
  const float* bk = (const float*)d_in[11];
  const float* Wv = (const float*)d_in[12];
  const float* bv = (const float*)d_in[13];
  const float* Wo = (const float*)d_in[14];
  const float* bo = (const float*)d_in[15];
  const float* gm = (const float*)d_in[16];
  const float* bt = (const float*)d_in[17];

  const size_t MB = 1024 * 1024;
  char* ws = (char*)d_ws;
  _Float16* qh   = (_Float16*)(ws);            //  4 MB : Q  (rope'd, row-major)
  _Float16* kh   = (_Float16*)(ws + 4 * MB);   // 16 MB : K  (rope'd, row-major)
  _Float16* vt   = (_Float16*)(ws + 20 * MB);  // 16 MB : V^T
  _Float16* pxw  = (_Float16*)(ws + 36 * MB);  //  4 MB : attention output
  _Float16* wq16 = (_Float16*)(ws + 40 * MB);  //  1 MB each
  _Float16* wk16 = (_Float16*)(ws + 41 * MB);
  _Float16* wv16 = (_Float16*)(ws + 42 * MB);
  _Float16* wo16 = (_Float16*)(ws + 43 * MB);
  _Float16* t16  = (_Float16*)(ws + 44 * MB);  //  2 MB : tgt f16
  _Float16* m16  = (_Float16*)(ws + 46 * MB);  //  8 MB : mem f16

  const int NW = HK * HID;  // 524288 weight elems
  cvt_kernel<<<NW / (256 * 8), 256, 0, stream>>>(Wq, wq16, NW);
  cvt_kernel<<<NW / (256 * 8), 256, 0, stream>>>(Wk, wk16, NW);
  cvt_kernel<<<NW / (256 * 8), 256, 0, stream>>>(Wv, wv16, NW);
  cvt_kernel<<<NW / (256 * 8), 256, 0, stream>>>(Wo, wo16, NW);
  const int NT = BB * NN * HID;   // 1048576
  const int NM = BB * MM * HID;   // 4194304
  cvt_kernel<<<NT / (256 * 8), 256, 0, stream>>>(tgt, t16, NT);
  cvt_kernel<<<NM / (256 * 8), 256, 0, stream>>>(mem, m16, NM);

  // Q: 2048 waves (nRT=128) -> 256 blocks; K/V: 8192 waves (nRT=512) -> 1024
  proj_kernel<<<256, 256, 0, stream>>>(t16, wq16, bq, gly, qh, NN, BB * NN / 16, 0);
  proj_kernel<<<1024, 256, 0, stream>>>(m16, wk16, bk, nme, kh, MM, BB * MM / 16, 0);
  proj_kernel<<<1024, 256, 0, stream>>>(m16, wv16, bv, nullptr, vt, MM, BB * MM / 16, 1);
  // attention: B*H*(N/16) = 2048 waves -> 512 blocks of 4 waves
  attn_kernel<<<512, 128, 0, stream>>>(qh, kh, vt, mask, pxw);
  // out-proj + residual + LN: 128 blocks of 16 rows
  out_ln_kernel<<<128, 256, 0, stream>>>(pxw, wo16, bo, tgt, gm, bt, (float*)d_out);
  // pass-through second output A (1 int32) at the tail of d_out
  hipMemcpyAsync((float*)d_out + (out_size - 1), A, 4,
                 hipMemcpyDeviceToDevice, stream);
}
